// Mamba_82420422410417
// MI455X (gfx1250) — matrix operand order repaired and checked by reading, not compiled, not run on hardware
//
#include <hip/hip_runtime.h>

typedef __attribute__((ext_vector_type(16))) _Float16 v16h;
typedef __attribute__((ext_vector_type(8)))  float    v8f;

#define B_   4
#define S_   2048
#define DM   1024
#define DI   2048
#define NST  16
#define MTOT (B_ * S_)   // 8192 tokens

// ---- CDNA5 async global->LDS copy (ASYNCcnt path, no VGPR staging) ----
__device__ __forceinline__ void async_b128(unsigned lds_addr, const void* gaddr) {
  asm volatile("global_load_async_to_lds_b128 %0, %1, off"
               :: "v"(lds_addr), "v"((unsigned long long)(size_t)gaddr)
               : "memory");
}
__device__ __forceinline__ unsigned lds_lo32(const void* p) {
  // LDS aperture: flat addr[31:0] == LDS byte offset
  return (unsigned)(size_t)p;
}

// -------- weight convert + transpose: f32 [K,N] row-major -> f16 [N,K] row-major --------
__global__ __launch_bounds__(256) void cvt_t_kernel(const float* __restrict__ W,
                                                    _Float16* __restrict__ Wt,
                                                    int K, int N) {
  size_t idx = (size_t)blockIdx.x * 256 + threadIdx.x;
  if (idx >= (size_t)K * N) return;
  int k = (int)(idx / N), n = (int)(idx % N);
  Wt[(size_t)n * K + k] = (_Float16)W[idx];  // coalesced read, scattered one-time write
}

// -------- LayerNorm (biased variance) -> f16 activations --------
__global__ __launch_bounds__(256) void ln_kernel(const float* __restrict__ x,
                                                 const float* __restrict__ g,
                                                 const float* __restrict__ b,
                                                 _Float16* __restrict__ xh) {
  int row = blockIdx.x;                       // 8192 rows of 1024
  const float* xr = x + (size_t)row * DM;
  float v[4], s = 0.f, s2 = 0.f;
#pragma unroll
  for (int i = 0; i < 4; ++i) {
    v[i] = xr[threadIdx.x + i * 256];
    s += v[i]; s2 += v[i] * v[i];
  }
#pragma unroll
  for (int off = 16; off; off >>= 1) {        // wave32 butterfly
    s  += __shfl_xor(s, off);
    s2 += __shfl_xor(s2, off);
  }
  __shared__ float rs[8], rs2[8];
  int wid = threadIdx.x >> 5;
  if ((threadIdx.x & 31) == 0) { rs[wid] = s; rs2[wid] = s2; }
  __syncthreads();
  s = 0.f; s2 = 0.f;
#pragma unroll
  for (int i = 0; i < 8; ++i) { s += rs[i]; s2 += rs2[i]; }
  float mu  = s * (1.f / DM);
  float var = s2 * (1.f / DM) - mu * mu;
  float r   = rsqrtf(var + 1e-5f);
#pragma unroll
  for (int i = 0; i < 4; ++i) {
    int col = threadIdx.x + i * 256;
    xh[(size_t)row * DM + col] = (_Float16)((v[i] - mu) * r * g[col] + b[col]);
  }
}

// -------- tiled WMMA GEMM, async double-buffered LDS pipeline (peeled drain) --------
// out[M,N] = A[M,K](f16) * Bt[N,K](f16)^T, fused epilogues:
// MODE 0: out0 = acc + bias                      (u = xn @ W_up + b_up)
// MODE 1: q = sigmoid(silu(acc+bias)); split cols into out0 (delta) / out1 (gating)
// MODE 2: out0 = acc + bias + resid              (down proj + residual)
template <int MODE>
__global__ __launch_bounds__(256) void gemm_wmma(const _Float16* __restrict__ Ag,
                                                 const _Float16* __restrict__ Bt,
                                                 const float* __restrict__ bias,
                                                 const float* __restrict__ resid,
                                                 float* __restrict__ out0,
                                                 float* __restrict__ out1,
                                                 int K, int N) {
  constexpr int BK  = 64;                     // K-slab per pipeline stage
  constexpr int LDT = 72;                     // padded LDS row stride in halves (144B)
  __shared__ _Float16 As[2][128 * LDT];
  __shared__ _Float16 Bs[2][128 * LDT];

  const int m0 = blockIdx.x * 128;
  const int n0 = blockIdx.y * 128;
  const int t = threadIdx.x;
  const int wid = t >> 5, lane = t & 31;
  const int waveM = wid >> 1, waveN = wid & 1;  // 4x2 wave grid, wave tile 32x64
  const int lhalf = lane & 15;
  const bool hi = lane >= 16;

  const int srow = t >> 1;                    // staging: 2 threads/row, 32 halves each
  const int sseg = (t & 1) * 32;

  const _Float16* aSrc = Ag + (size_t)(m0 + srow) * K + sseg;
  const _Float16* bSrc = Bt + (size_t)(n0 + srow) * K + sseg;
  const unsigned aLds0 = lds_lo32(&As[0][srow * LDT + sseg]);
  const unsigned bLds0 = lds_lo32(&Bs[0][srow * LDT + sseg]);
  const unsigned bufStride = (unsigned)(128 * LDT * sizeof(_Float16));

  v8f zero = {0.f, 0.f, 0.f, 0.f, 0.f, 0.f, 0.f, 0.f};
  v8f acc[2][4];
#pragma unroll
  for (int i = 0; i < 2; ++i)
#pragma unroll
    for (int j = 0; j < 4; ++j) acc[i][j] = zero;

  // issue 8 async b128 copies (64B of A + 64B of B per thread) for one stage
  auto issueTile = [&](int buf, int k0) {
    const _Float16* ga = aSrc + k0;
    const _Float16* gb = bSrc + k0;
    unsigned la = aLds0 + buf * bufStride;
    unsigned lb = bLds0 + buf * bufStride;
#pragma unroll
    for (int j = 0; j < 4; ++j) {
      async_b128(la + j * 16, ga + j * 8);
      async_b128(lb + j * 16, gb + j * 8);
    }
  };

  // consume one BK slab from LDS buffer `buf` (2 k-substeps, 16 WMMAs)
  auto computeSlab = [&](int buf) {
#pragma unroll
    for (int ks = 0; ks < BK; ks += 32) {
      // A frag (16-bit 16x32, ISA 7.12.2): lo lanes K={0..7,16..23}, hi K={8..15,24..31}
      v16h af[2], bf[4];
#pragma unroll
      for (int mt = 0; mt < 2; ++mt) {
        const _Float16* p =
            &As[buf][(waveM * 32 + mt * 16 + lhalf) * LDT + ks + (hi ? 8 : 0)];
        ((uint4*)&af[mt])[0] = *(const uint4*)(p);
        ((uint4*)&af[mt])[1] = *(const uint4*)(p + 16);
      }
      // B frag: lo lanes K={0..7,16..23}, hi K={8..15,24..31} (same order as A)
#pragma unroll
      for (int nt = 0; nt < 4; ++nt) {
        const _Float16* p =
            &Bs[buf][(waveN * 64 + nt * 16 + lhalf) * LDT + ks + (hi ? 8 : 0)];
        ((uint4*)&bf[nt])[0] = *(const uint4*)(p);
        ((uint4*)&bf[nt])[1] = *(const uint4*)(p + 16);
      }
#pragma unroll
      for (int mt = 0; mt < 2; ++mt)
#pragma unroll
        for (int nt = 0; nt < 4; ++nt)
          acc[mt][nt] = __builtin_amdgcn_wmma_f32_16x16x32_f16(
              false, af[mt], false, bf[nt], (short)0, acc[mt][nt], false, false);
    }
  };

  const int nIter = K / BK;
  issueTile(0, 0);                            // prologue: fill stage 0

  // steady state: branch-free issue -> partial-wait -> barrier -> compute -> barrier
  for (int it = 0; it < nIter - 1; ++it) {
    const int buf = it & 1;
    issueTile(buf ^ 1, (it + 1) * BK);        // stream next slab behind compute
    asm volatile("s_wait_asynccnt 0x8" ::: "memory");  // current slab's 8 ops done
    __syncthreads();
    computeSlab(buf);
    __syncthreads();                          // all waves done reading buf before reuse
  }

  // drain: last slab, no further issue
  asm volatile("s_wait_asynccnt 0x0" ::: "memory");
  __syncthreads();
  computeSlab((nIter - 1) & 1);

  // epilogue: C/D layout (ISA 7.12.2): VGPR r -> M = r + (hi?8:0), N = lane%16
#pragma unroll
  for (int mt = 0; mt < 2; ++mt) {
#pragma unroll
    for (int nt = 0; nt < 4; ++nt) {
      int rowb = m0 + waveM * 32 + mt * 16 + (hi ? 8 : 0);
      int col  = n0 + waveN * 64 + nt * 16 + lhalf;
#pragma unroll
      for (int r = 0; r < 8; ++r) {
        int row = rowb + r;
        float vv = acc[mt][nt][r];
        if (MODE == 0) {
          out0[(size_t)row * N + col] = vv + bias[col];
        } else if (MODE == 1) {
          float z  = vv + bias[col];
          float sl = z / (1.f + __expf(-z));        // silu
          float q  = 1.f / (1.f + __expf(-sl));     // sigmoid (scan uses it directly)
          int half = N >> 1;
          if (col < half) out0[(size_t)row * half + col] = q;
          else            out1[(size_t)row * half + (col - half)] = q;
        } else {
          out0[(size_t)row * N + col] = vv + bias[col] + resid[(size_t)row * N + col];
        }
      }
    }
  }
}

// -------- selective-scan: one thread per (batch, channel), 16 states in registers --------
__global__ __launch_bounds__(256) void scan_kernel(const float* __restrict__ u,
                                                   const float* __restrict__ dt,
                                                   const float* __restrict__ gs,
                                                   const float* __restrict__ Am,
                                                   const float* __restrict__ Bm,
                                                   const float* __restrict__ Cm,
                                                   const float* __restrict__ Dv,
                                                   _Float16* __restrict__ yh) {
  int c = blockIdx.x * 256 + threadIdx.x;     // 0..8191 = b*DI + ch
  int b = c >> 11, ch = c & (DI - 1);
  float a[NST], bb[NST], cc[NST], h[NST];
#pragma unroll
  for (int n = 0; n < NST; ++n) {
    a[n]  = Am[ch * NST + n];
    bb[n] = Bm[ch * NST + n];
    cc[n] = Cm[ch * NST + n];
    h[n]  = 0.f;
  }
  float dv = Dv[ch];
  size_t idx = (size_t)b * S_ * DI + ch;      // coalesced across ch per step
  for (int tstep = 0; tstep < S_; ++tstep, idx += DI) {
    float uu = u[idx], d = dt[idx], sg = gs[idx];
    float y = uu * dv;
#pragma unroll
    for (int n = 0; n < NST; ++n) {
      h[n] = h[n] * __expf(d * a[n]) + uu * bb[n];
      y += h[n] * cc[n];
    }
    yh[idx] = (_Float16)(y * sg);
  }
}

extern "C" void kernel_launch(void* const* d_in, const int* in_sizes, int n_in,
                              void* d_out, int out_size, void* d_ws, size_t ws_size,
                              hipStream_t stream) {
  const float* x    = (const float*)d_in[0];
  const float* ln_g = (const float*)d_in[1];
  const float* ln_b = (const float*)d_in[2];
  const float* W_up = (const float*)d_in[3];
  const float* b_up = (const float*)d_in[4];
  const float* Am   = (const float*)d_in[5];
  const float* Bm   = (const float*)d_in[6];
  const float* Cm   = (const float*)d_in[7];
  const float* Dv   = (const float*)d_in[8];
  const float* W_p  = (const float*)d_in[9];
  const float* b_p  = (const float*)d_in[10];
  const float* W_dn = (const float*)d_in[11];
  const float* b_dn = (const float*)d_in[12];
  float* out = (float*)d_out;

  char* ws = (char*)d_ws;
  const size_t MB = 1024ull * 1024ull;
  _Float16* xh   = (_Float16*)(ws + 0);        // 16 MiB  [8192,1024] f16
  _Float16* WupT = (_Float16*)(ws + 16 * MB);  //  4 MiB  [2048,1024] f16
  _Float16* WpT  = (_Float16*)(ws + 20 * MB);  //  8 MiB  [4096,1024] f16
  _Float16* WdnT = (_Float16*)(ws + 28 * MB);  //  4 MiB  [1024,2048] f16
  float*    u    = (float*)   (ws + 32 * MB);  // 64 MiB  [8192,2048] f32
  float*    dt   = (float*)   (ws + 96 * MB);  // 64 MiB
  float*    g    = (float*)   (ws + 160 * MB); // 64 MiB
  _Float16* yh   = (_Float16*)(ws + 224 * MB); // 32 MiB  [8192,2048] f16

  cvt_t_kernel<<<(1024 * 2048) / 256, 256, 0, stream>>>(W_up, WupT, 1024, 2048);
  cvt_t_kernel<<<(1024 * 4096) / 256, 256, 0, stream>>>(W_p,  WpT,  1024, 4096);
  cvt_t_kernel<<<(2048 * 1024) / 256, 256, 0, stream>>>(W_dn, WdnT, 2048, 1024);

  ln_kernel<<<MTOT, 256, 0, stream>>>(x, ln_g, ln_b, xh);

  gemm_wmma<0><<<dim3(MTOT / 128, 2048 / 128), 256, 0, stream>>>(
      xh, WupT, b_up, nullptr, u, nullptr, 1024, 2048);
  gemm_wmma<1><<<dim3(MTOT / 128, 4096 / 128), 256, 0, stream>>>(
      xh, WpT, b_p, nullptr, dt, g, 1024, 4096);

  scan_kernel<<<(B_ * DI) / 256, 256, 0, stream>>>(u, dt, g, Am, Bm, Cm, Dv, yh);

  gemm_wmma<2><<<dim3(MTOT / 128, 1024 / 128), 256, 0, stream>>>(
      yh, WdnT, b_dn, x, out, nullptr, 2048, 1024);
}
